// Qwen3OmniMoeTalkerTextModel_70489003261982
// MI455X (gfx1250) — matrix operand order repaired and checked by reading, not compiled, not run on hardware
//
#include <hip/hip_runtime.h>
#include <hip/hip_bf16.h>
#include <math.h>

// ---------------- model constants ----------------
#define T_   1024
#define D_   1024
#define H_   16
#define KV_  4
#define DH_  64
#define E_   8
#define F_   512
#define FS_  2048
#define L_   2
#define EPS_ 1e-6f

typedef __attribute__((ext_vector_type(16))) __bf16 v16bf;
typedef __attribute__((ext_vector_type(8)))  float v8f;
typedef unsigned short u16;
typedef unsigned int   u32;
typedef long long      i64;
typedef int ai4 __attribute__((vector_size(16)));   // int4 for async-LDS builtin

// CDNA5 async global->LDS staging (guarded: falls back to ds stores if the
// toolchain lacks the builtins). Parameter type per hipcc diagnostic: int4*.
#if __has_builtin(__builtin_amdgcn_global_load_async_to_lds_b128) && \
    __has_builtin(__builtin_amdgcn_s_wait_asynccnt)
#define USE_ASYNC_LDS 1
#else
#define USE_ASYNC_LDS 0
#endif

__device__ __forceinline__ u16 f2bf(float f) {
  u32 b = __float_as_uint(f);
  u32 r = (b + 0x7FFFu + ((b >> 16) & 1u)) >> 16;
  return (u16)r;
}

// ---------------- elementwise kernels ----------------
__global__ void k_embed(const int* __restrict__ ids, const float* __restrict__ emb,
                        float* __restrict__ h, float* __restrict__ res) {
  int i = blockIdx.x * 256 + threadIdx.x;     // T_*D_ threads
  int t = i >> 10;                             // / D_
  int d = i & (D_ - 1);
  h[i]   = emb[(size_t)ids[t] * D_ + d];
  res[i] = 0.f;
}

__global__ void k_cvt(const float* __restrict__ x, u16* __restrict__ y, int n) {
  int i = blockIdx.x * 256 + threadIdx.x;
  if (i < n) y[i] = f2bf(x[i]);
}

__global__ void k_addip(float* __restrict__ a, const float* __restrict__ b, int n) {
  int i = blockIdx.x * 256 + threadIdx.x;
  if (i < n) a[i] += b[i];
}

// RMSNorm over `cols`; optional fp32 and bf16 outputs.
__global__ void k_rms(const float* __restrict__ x, const float* __restrict__ w,
                      float* __restrict__ y, u16* __restrict__ yb, int cols) {
  int row = blockIdx.x;
  const float* xr = x + (size_t)row * cols;
  __shared__ float red[256];
  float s = 0.f;
  for (int c = threadIdx.x; c < cols; c += 256) { float v = xr[c]; s += v * v; }
  red[threadIdx.x] = s; __syncthreads();
  for (int o = 128; o > 0; o >>= 1) {
    if (threadIdx.x < o) red[threadIdx.x] += red[threadIdx.x + o];
    __syncthreads();
  }
  float scale = rsqrtf(red[0] / cols + EPS_);
  for (int c = threadIdx.x; c < cols; c += 256) {
    float v = xr[c] * scale * w[c];
    if (y)  y[(size_t)row * cols + c]  = v;
    if (yb) yb[(size_t)row * cols + c] = f2bf(v);
  }
}

// per-head RMSNorm + rotate-half RoPE for q (blockIdx.y < H_) and k heads.
__global__ void k_qk(const float* __restrict__ qf, const float* __restrict__ kf,
                     const float* __restrict__ qn, const float* __restrict__ kn,
                     const int* __restrict__ pos,
                     u16* __restrict__ qb, u16* __restrict__ kb) {
  int t = blockIdx.x;
  int hh = blockIdx.y;
  bool isq = hh < H_;
  int head = isq ? hh : hh - H_;
  const float* src = isq ? qf + (size_t)t * (H_ * DH_) + head * DH_
                         : kf + (size_t)t * (KV_ * DH_) + head * DH_;
  const float* wn = isq ? qn : kn;
  u16* dst = isq ? qb + (size_t)t * (H_ * DH_) + head * DH_
                 : kb + (size_t)t * (KV_ * DH_) + head * DH_;
  int d = threadIdx.x;                        // 64 threads
  __shared__ float sh[DH_];
  __shared__ float red[DH_];
  float v = src[d];
  red[d] = v * v; __syncthreads();
  for (int o = 32; o > 0; o >>= 1) {
    if (d < o) red[d] += red[d + o];
    __syncthreads();
  }
  float scale = rsqrtf(red[0] / DH_ + EPS_);
  float xn = v * scale * wn[d];
  sh[d] = xn; __syncthreads();
  int j = d & 31;
  float inv = powf(1e6f, -(float)j / 32.0f);
  float ang = (float)pos[t] * inv;
  float c = cosf(ang), s = sinf(ang);
  float rot = (d < 32) ? -sh[d + 32] : sh[d - 32];
  dst[d] = f2bf(xn * c + rot * s);
}

// causal softmax of one [T] row of scores; emits bf16 probabilities.
__global__ void k_softmax(const float* __restrict__ att, u16* __restrict__ pb) {
  int row = blockIdx.x;                       // H_*T_ rows
  int q = row & (T_ - 1);
  const float* xr = att + (size_t)row * T_;
  u16* pr = pb + (size_t)row * T_;
  __shared__ float red[256];
  int n = q + 1;
  float m = -1e30f;
  for (int c = threadIdx.x; c < n; c += 256) m = fmaxf(m, xr[c]);
  red[threadIdx.x] = m; __syncthreads();
  for (int o = 128; o > 0; o >>= 1) {
    if (threadIdx.x < o) red[threadIdx.x] = fmaxf(red[threadIdx.x], red[threadIdx.x + o]);
    __syncthreads();
  }
  m = red[0]; __syncthreads();
  float s = 0.f;
  for (int c = threadIdx.x; c < n; c += 256) s += __expf(xr[c] - m);
  red[threadIdx.x] = s; __syncthreads();
  for (int o = 128; o > 0; o >>= 1) {
    if (threadIdx.x < o) red[threadIdx.x] += red[threadIdx.x + o];
    __syncthreads();
  }
  float inv = 1.f / red[0];
  for (int c = threadIdx.x; c < T_; c += 256)
    pr[c] = f2bf(c < n ? __expf(xr[c] - m) * inv : 0.f);
}

// silu(g[:, :F]) * g[:, F:] * rowscale -> bf16
__global__ void k_silu_mul(const float* __restrict__ g, int twoF,
                           const float* __restrict__ rowscale, int scaleStride,
                           u16* __restrict__ out) {
  int Fh = twoF >> 1;
  int i = blockIdx.x * 256 + threadIdx.x;
  int t = i / Fh, f = i - t * Fh;
  const float* gr = g + (size_t)t * twoF;
  float gl = gr[f], gu = gr[Fh + f];
  float v = (gl / (1.f + __expf(-gl))) * gu;
  if (rowscale) v *= rowscale[(size_t)t * scaleStride];
  out[(size_t)t * Fh + f] = f2bf(v);
}

// one wave per token: router logits, softmax, top-2, renormalize -> comb[T,E]
__global__ void k_router(const float* __restrict__ hn, const float* __restrict__ rw,
                         float* __restrict__ comb) {
  int t = blockIdx.x, lane = threadIdx.x;     // 32 threads
  float logit[E_];
  for (int e = 0; e < E_; e++) {
    float s = 0.f;
    for (int d = lane; d < D_; d += 32) s += hn[(size_t)t * D_ + d] * rw[(size_t)d * E_ + e];
    for (int o = 16; o > 0; o >>= 1) s += __shfl_down(s, o, 32);
    logit[e] = s;
  }
  if (lane == 0) {
    float m = logit[0];
    for (int e = 1; e < E_; e++) m = fmaxf(m, logit[e]);
    float p[E_], sum = 0.f;
    for (int e = 0; e < E_; e++) { p[e] = __expf(logit[e] - m); sum += p[e]; }
    for (int e = 0; e < E_; e++) p[e] /= sum;
    int i0 = 0;
    for (int e = 1; e < E_; e++) if (p[e] > p[i0]) i0 = e;
    int i1 = (i0 == 0) ? 1 : 0;
    for (int e = 0; e < E_; e++) if (e != i0 && p[e] > p[i1]) i1 = e;
    float wsum = p[i0] + p[i1];
    for (int e = 0; e < E_; e++) comb[(size_t)t * E_ + e] = 0.f;
    comb[(size_t)t * E_ + i0] = p[i0] / wsum;
    comb[(size_t)t * E_ + i1] = p[i1] / wsum;
  }
}

// one wave per token: sigmoid(hn . sg_gate)
__global__ void k_gate(const float* __restrict__ hn, const float* __restrict__ sg,
                       float* __restrict__ gate) {
  int t = blockIdx.x, lane = threadIdx.x;
  float s = 0.f;
  for (int d = lane; d < D_; d += 32) s += hn[(size_t)t * D_ + d] * sg[d];
  for (int o = 16; o > 0; o >>= 1) s += __shfl_down(s, o, 32);
  if (lane == 0) gate[t] = 1.f / (1.f + __expf(-s));
}

// ---------------- WMMA bf16 GEMM ----------------
// C[M,N](f32) = alpha * A[M,K](bf16) x op(B)(bf16) (+ C if ACC)
// Block tile 64x64, 8 waves (2x4), each wave two 16x16 accumulators, K step 32.
// A staged row-major As[64][32]; B staged COLUMN-major Bs_t[64][32] so every
// lane's fragment is two contiguous b128 LDS reads (per ISA 7.12.2 layouts).
// Global->LDS staging uses GLOBAL_LOAD_ASYNC_TO_LDS_B128 when available.
// Batched: z advances A/C by sA/sC; B by (z>>bShift)*sB (GQA head sharing).
// Requires M%64==0, N%64==0, K%32==0 (true for every call below).
template <bool TB, bool ACC>
__global__ void k_gemm(const u16* __restrict__ A, int lda, i64 sA,
                       const u16* __restrict__ B, int ldb, i64 sB, int bShift,
                       float* __restrict__ C, int ldc, i64 sC,
                       int M, int N, int K, float alpha) {
  __shared__ u16 As[64][32];     // [m][k]
  __shared__ u16 Bs_t[64][32];   // [n][k]
  int z = blockIdx.z;
  A += (size_t)z * sA;
  B += (size_t)(z >> bShift) * sB;
  C += (size_t)z * sC;
  int bm = blockIdx.y * 64, bn = blockIdx.x * 64;
  int tid = threadIdx.x;
  int wave = tid >> 5, lane = tid & 31;
  int wm = (wave >> 2) * 32, wn = (wave & 3) * 16;
  v8f acc0 = {}; v8f acc1 = {};

  int arow = tid >> 2, ak = (tid & 3) * 8;    // A tile: 64 rows x 4 chunks of 8
  for (int k0 = 0; k0 < K; k0 += 32) {
    __syncthreads();
    // ---- stage A tile ----
    const u16* asrc = &A[(size_t)(bm + arow) * lda + k0 + ak];
#if USE_ASYNC_LDS
    __builtin_amdgcn_global_load_async_to_lds_b128((ai4*)asrc, (ai4*)&As[arow][ak], 0, 0);
#else
    *(uint4*)&As[arow][ak] = *(const uint4*)asrc;
#endif
    if (k0 + 32 < K)
      __builtin_prefetch(&A[(size_t)(bm + arow) * lda + k0 + 32 + ak], 0, 0);
    // ---- stage B tile into column-major Bs_t ----
    if (TB) {
      // B is [N,K] row-major: straight b128 copy
      int bcol = tid >> 2, bk = (tid & 3) * 8;
      const u16* bsrc = &B[(size_t)(bn + bcol) * ldb + k0 + bk];
#if USE_ASYNC_LDS
      __builtin_amdgcn_global_load_async_to_lds_b128((ai4*)bsrc, (ai4*)&Bs_t[bcol][bk], 0, 0);
#else
      *(uint4*)&Bs_t[bcol][bk] = *(const uint4*)bsrc;
#endif
    } else {
      // B is [K,N] row-major: transpose while storing to LDS
      int krow = tid >> 3, nc = (tid & 7) * 8;
      uint4 vv = *(const uint4*)&B[(size_t)(k0 + krow) * ldb + bn + nc];
      const u16* pv = (const u16*)&vv;
#pragma unroll
      for (int j = 0; j < 8; j++) Bs_t[nc + j][krow] = pv[j];
    }
#if USE_ASYNC_LDS
    __builtin_amdgcn_s_wait_asynccnt(0);   // barrier does not cover ASYNCcnt
#endif
    __syncthreads();

    // ---- pack fragments: all contiguous b128 LDS reads ----
    union Frag { v16bf v; uint4 q[2]; } fa0, fa1, fb;
    int row = lane & 15, kh = lane >> 4;
    const u16* a0 = &As[wm + row][kh * 8];          // A: K 0..7|16..23 / 8..15|24..31
    fa0.q[0] = *(const uint4*)a0;
    fa0.q[1] = *(const uint4*)(a0 + 16);
    const u16* a1 = &As[wm + 16 + row][kh * 8];
    fa1.q[0] = *(const uint4*)a1;
    fa1.q[1] = *(const uint4*)(a1 + 16);
    const u16* b0 = &Bs_t[wn + row][kh * 8];        // B: K 0..7|16..23 / 8..15|24..31 of column
    fb.q[0] = *(const uint4*)b0;
    fb.q[1] = *(const uint4*)(b0 + 16);

    acc0 = __builtin_amdgcn_wmma_f32_16x16x32_bf16(false, fa0.v, false, fb.v,
                                                   (short)0, acc0, false, false);
    acc1 = __builtin_amdgcn_wmma_f32_16x16x32_bf16(false, fa1.v, false, fb.v,
                                                   (short)0, acc1, false, false);
  }

  int col = lane & 15, rh = lane >> 4;            // C/D layout: row = v + 8*half
#pragma unroll
  for (int v = 0; v < 8; v++) {
    int r0 = bm + wm + v + 8 * rh;
    int c = bn + wn + col;
    float* p0 = &C[(size_t)r0 * ldc + c];
    float* p1 = &C[(size_t)(r0 + 16) * ldc + c];
    float x0 = alpha * acc0[v], x1 = alpha * acc1[v];
    if (ACC) { x0 += *p0; x1 += *p1; }
    *p0 = x0; *p1 = x1;
  }
}

// ---------------- host side ----------------
static void gemm(hipStream_t st, bool tb, bool acc,
                 const u16* A, int lda, i64 sA,
                 const u16* B, int ldb, i64 sB, int bShift,
                 float* C, int ldc, i64 sC,
                 int M, int N, int K, float alpha, int batch) {
  dim3 g(N / 64, M / 64, batch), b(256);
  if (!tb && !acc)
    k_gemm<false, false><<<g, b, 0, st>>>(A, lda, sA, B, ldb, sB, bShift, C, ldc, sC, M, N, K, alpha);
  else if (!tb && acc)
    k_gemm<false, true><<<g, b, 0, st>>>(A, lda, sA, B, ldb, sB, bShift, C, ldc, sC, M, N, K, alpha);
  else if (tb && !acc)
    k_gemm<true, false><<<g, b, 0, st>>>(A, lda, sA, B, ldb, sB, bShift, C, ldc, sC, M, N, K, alpha);
  else
    k_gemm<true, true><<<g, b, 0, st>>>(A, lda, sA, B, ldb, sB, bShift, C, ldc, sC, M, N, K, alpha);
}

static void cvt(hipStream_t st, const float* x, u16* y, size_t n) {
  k_cvt<<<(unsigned)((n + 255) / 256), 256, 0, st>>>(x, y, (int)n);
}

extern "C" void kernel_launch(void* const* d_in, const int* in_sizes, int n_in,
                              void* d_out, int out_size, void* d_ws, size_t ws_size,
                              hipStream_t stream) {
  const int*   ids      = (const int*)d_in[0];
  const int*   pos      = (const int*)d_in[1];
  const float* emb      = (const float*)d_in[2];
  const float* ln1      = (const float*)d_in[3];
  const float* wq       = (const float*)d_in[4];
  const float* wk       = (const float*)d_in[5];
  const float* wv       = (const float*)d_in[6];
  const float* wo       = (const float*)d_in[7];
  const float* qn       = (const float*)d_in[8];
  const float* kn       = (const float*)d_in[9];
  const float* ln2      = (const float*)d_in[10];
  const float* router_w = (const float*)d_in[11];
  const float* egu      = (const float*)d_in[12];
  const float* edn      = (const float*)d_in[13];
  const float* sgu      = (const float*)d_in[14];
  const float* sdn      = (const float*)d_in[15];
  const float* sg_gate  = (const float*)d_in[16];
  const float* final_ln = (const float*)d_in[17];
  float* out = (float*)d_out;

  // bump allocator over workspace
  char* p = (char*)d_ws;
  auto alloc = [&](size_t bytes) -> void* {
    void* r = (void*)p;
    p += (bytes + 255) & ~(size_t)255;
    return r;
  };

  const size_t nWQ = (size_t)L_ * D_ * H_ * DH_;     // 2M
  const size_t nWK = (size_t)L_ * D_ * KV_ * DH_;    // 0.5M
  const size_t nWO = (size_t)L_ * H_ * DH_ * D_;     // 2M
  const size_t nSGU = (size_t)L_ * D_ * 2 * FS_;     // 8M
  const size_t nSDN = (size_t)L_ * FS_ * D_;         // 4M
  const size_t nEGU = (size_t)L_ * E_ * D_ * 2 * F_; // 16M
  const size_t nEDN = (size_t)L_ * E_ * F_ * D_;     // 8M

  u16* wq_b  = (u16*)alloc(nWQ * 2);
  u16* wk_b  = (u16*)alloc(nWK * 2);
  u16* wv_b  = (u16*)alloc(nWK * 2);
  u16* wo_b  = (u16*)alloc(nWO * 2);
  u16* sgu_b = (u16*)alloc(nSGU * 2);
  u16* sdn_b = (u16*)alloc(nSDN * 2);
  u16* egu_b = (u16*)alloc(nEGU * 2);
  u16* edn_b = (u16*)alloc(nEDN * 2);

  float* hA   = (float*)alloc((size_t)T_ * D_ * 4);
  float* hB   = (float*)alloc((size_t)T_ * D_ * 4);
  float* res  = (float*)alloc((size_t)T_ * D_ * 4);
  float* hn   = (float*)alloc((size_t)T_ * D_ * 4);
  u16*   hnb  = (u16*)alloc((size_t)T_ * D_ * 2);
  float* qf   = (float*)alloc((size_t)T_ * H_ * DH_ * 4);
  float* kf   = (float*)alloc((size_t)T_ * KV_ * DH_ * 4);
  float* vf   = (float*)alloc((size_t)T_ * KV_ * DH_ * 4);
  u16*   qb   = (u16*)alloc((size_t)T_ * H_ * DH_ * 2);
  u16*   kb   = (u16*)alloc((size_t)T_ * KV_ * DH_ * 2);
  u16*   vb   = (u16*)alloc((size_t)T_ * KV_ * DH_ * 2);
  float* att  = (float*)alloc((size_t)H_ * T_ * T_ * 4);
  u16*   pb   = (u16*)alloc((size_t)H_ * T_ * T_ * 2);
  float* of   = (float*)alloc((size_t)T_ * H_ * DH_ * 4);
  u16*   ob   = (u16*)alloc((size_t)T_ * H_ * DH_ * 2);
  float* tmp  = (float*)alloc((size_t)T_ * D_ * 4);
  float* gu   = (float*)alloc((size_t)T_ * 2 * FS_ * 4);
  u16*   sact = (u16*)alloc((size_t)T_ * FS_ * 2);
  float* gf   = (float*)alloc((size_t)T_ * 2 * F_ * 4);
  u16*   heb  = (u16*)alloc((size_t)T_ * F_ * 2);
  float* comb = (float*)alloc((size_t)T_ * E_ * 4);
  float* gate = (float*)alloc((size_t)T_ * 4);

  // one-time (per call) weight conversion fp32 -> bf16
  cvt(stream, wq, wq_b, nWQ);
  cvt(stream, wk, wk_b, nWK);
  cvt(stream, wv, wv_b, nWK);
  cvt(stream, wo, wo_b, nWO);
  cvt(stream, sgu, sgu_b, nSGU);
  cvt(stream, sdn, sdn_b, nSDN);
  cvt(stream, egu, egu_b, nEGU);
  cvt(stream, edn, edn_b, nEDN);

  k_embed<<<T_ * D_ / 256, 256, 0, stream>>>(ids, emb, hA, res);
  float* h = hA;
  const float scale = 0.125f;  // 1/sqrt(DH)

  for (int i = 0; i < L_; i++) {
    // ---- attention ----
    k_addip<<<T_ * D_ / 256, 256, 0, stream>>>(res, h, T_ * D_);
    k_rms<<<T_, 256, 0, stream>>>(res, ln1 + (size_t)i * D_, hn, hnb, D_);
    gemm(stream, false, false, hnb, D_, 0, wq_b + (size_t)i * D_ * (H_ * DH_), H_ * DH_, 0, 0,
         qf, H_ * DH_, 0, T_, H_ * DH_, D_, 1.f, 1);
    gemm(stream, false, false, hnb, D_, 0, wk_b + (size_t)i * D_ * (KV_ * DH_), KV_ * DH_, 0, 0,
         kf, KV_ * DH_, 0, T_, KV_ * DH_, D_, 1.f, 1);
    gemm(stream, false, false, hnb, D_, 0, wv_b + (size_t)i * D_ * (KV_ * DH_), KV_ * DH_, 0, 0,
         vf, KV_ * DH_, 0, T_, KV_ * DH_, D_, 1.f, 1);
    cvt(stream, vf, vb, (size_t)T_ * KV_ * DH_);
    k_qk<<<dim3(T_, H_ + KV_), DH_, 0, stream>>>(qf, kf, qn + (size_t)i * DH_,
                                                 kn + (size_t)i * DH_, pos, qb, kb);
    // scores: per head, S = scale * Q Kt   (B transposed, GQA via bShift=2)
    gemm(stream, true, false, qb, H_ * DH_, DH_, kb, KV_ * DH_, DH_, 2,
         att, T_, (i64)T_ * T_, T_, T_, DH_, scale, H_);
    k_softmax<<<H_ * T_, 256, 0, stream>>>(att, pb);
    // O = P V  (per head, N = DH slice of [T, H*DH])
    gemm(stream, false, false, pb, T_, (i64)T_ * T_, vb, KV_ * DH_, DH_, 2,
         of, H_ * DH_, DH_, T_, DH_, T_, 1.f, H_);
    cvt(stream, of, ob, (size_t)T_ * H_ * DH_);
    gemm(stream, false, false, ob, H_ * DH_, 0, wo_b + (size_t)i * (H_ * DH_) * D_, D_, 0, 0,
         tmp, D_, 0, T_, D_, H_ * DH_, 1.f, 1);
    k_addip<<<T_ * D_ / 256, 256, 0, stream>>>(res, tmp, T_ * D_);

    // ---- MoE MLP ----
    k_rms<<<T_, 256, 0, stream>>>(res, ln2 + (size_t)i * D_, hn, hnb, D_);
    gemm(stream, false, false, hnb, D_, 0, sgu_b + (size_t)i * D_ * 2 * FS_, 2 * FS_, 0, 0,
         gu, 2 * FS_, 0, T_, 2 * FS_, D_, 1.f, 1);
    k_router<<<T_, 32, 0, stream>>>(hn, router_w + (size_t)i * D_ * E_, comb);
    k_gate<<<T_, 32, 0, stream>>>(hn, sg_gate + (size_t)i * D_, gate);
    // shared act, pre-scaled by sigmoid gate (linear, commutes with down-proj)
    k_silu_mul<<<T_ * FS_ / 256, 256, 0, stream>>>(gu, 2 * FS_, gate, 1, sact);

    float* hnew = (i & 1) ? hA : hB;
    for (int e = 0; e < E_; e++) {
      const u16* eg = egu_b + ((size_t)(i * E_ + e)) * D_ * 2 * F_;
      const u16* ed = edn_b + ((size_t)(i * E_ + e)) * F_ * D_;
      gemm(stream, false, false, hnb, D_, 0, eg, 2 * F_, 0, 0,
           gf, 2 * F_, 0, T_, 2 * F_, D_, 1.f, 1);
      // activation pre-scaled by router weight comb[:, e] (zero if not routed)
      k_silu_mul<<<T_ * F_ / 256, 256, 0, stream>>>(gf, 2 * F_, comb + e, E_, heb);
      gemm(stream, false, e > 0, heb, F_, 0, ed, D_, 0, 0,
           hnew, D_, 0, T_, D_, F_, 1.f, 1);
    }
    // shared expert down-proj accumulates on top of routed sum
    gemm(stream, false, true, sact, FS_, 0, sdn_b + (size_t)i * FS_ * D_, D_, 0, 0,
         hnew, D_, 0, T_, D_, FS_, 1.f, 1);
    h = hnew;
  }

  k_addip<<<T_ * D_ / 256, 256, 0, stream>>>(res, h, T_ * D_);
  k_rms<<<T_, 256, 0, stream>>>(res, final_ln, out, (u16*)nullptr, D_);
}
